// PointNetpp_91122026152140
// MI455X (gfx1250) — compile-verified, matrix operand order checked by reading, not run on hardware
//
#include <hip/hip_runtime.h>
#include <hip/hip_bf16.h>

// PointNet++ part-segmentation forward for gfx1250 (MI455X).
// All MLP layers run through a branch-free WMMA f16->f32 GEMM
// (v_wmma_f32_16x16x32_f16) over K-padded f16 activations and
// pre-transposed f16 weights. Each wave computes a 32x32 output tile
// (2x2 WMMA tiles) for register-level operand reuse.

typedef __attribute__((ext_vector_type(16))) _Float16 v16h;
typedef __attribute__((ext_vector_type(8)))  _Float16 h8;
typedef __attribute__((ext_vector_type(8)))  float    v8f;

union V16 { v16h v; h8 h[2]; };

static constexpr int Bc  = 4;
static constexpr int Nc  = 16384;
static constexpr int NS  = 32;          // nsample
static constexpr int CLS = 13;

// ---------------------------------------------------------------------------
// Split input x(B,N,9) into c0(B,N,3) f32 and f0(B,N,6) f16
// ---------------------------------------------------------------------------
__global__ void split_kernel(const float* __restrict__ x, float* __restrict__ c0,
                             _Float16* __restrict__ f0, int total) {
  int t = blockIdx.x * blockDim.x + threadIdx.x;
  if (t >= total) return;
  int ch = t % 9;
  long p = t / 9;
  float v = x[t];
  if (ch < 3) c0[p * 3 + ch] = v;
  else        f0[p * 6 + (ch - 3)] = (_Float16)v;
}

// ---------------------------------------------------------------------------
// Weight convert + transpose: Wt(N,Kp) f16 = W(K,N) f32, zero pad K..Kp
// ---------------------------------------------------------------------------
__global__ void wconv_kernel(const float* __restrict__ W, _Float16* __restrict__ Wt,
                             int K, int Kp, int N, int total) {
  int t = blockIdx.x * blockDim.x + threadIdx.x;
  if (t >= total) return;
  int k = t % Kp;
  int n = t / Kp;
  Wt[t] = (k < K) ? (_Float16)W[(long)k * N + n] : (_Float16)0.0f;
}

// ---------------------------------------------------------------------------
// Farthest point sampling: one block per batch, iterative argmax.
// ---------------------------------------------------------------------------
#define FPS_T 256
__global__ void fps_kernel(const float* __restrict__ xyz, int* __restrict__ out,
                           float* __restrict__ dist, int Np, int np) {
  int b = blockIdx.x;
  const float* p = xyz + (long)b * Np * 3;
  float* db = dist + (long)b * Np;
  __shared__ float sval[FPS_T];
  __shared__ int   sidx[FPS_T];
  __shared__ int   sfar;
  for (int i = threadIdx.x; i < Np; i += FPS_T) db[i] = 1e10f;
  if (threadIdx.x == 0) sfar = 0;
  __syncthreads();
  for (int s = 0; s < np; ++s) {
    int far = sfar;
    if (threadIdx.x == 0) out[(long)b * np + s] = far;
    float cx = p[far * 3 + 0], cy = p[far * 3 + 1], cz = p[far * 3 + 2];
    float bv = -1.0f; int bi = 0;
    for (int i = threadIdx.x; i < Np; i += FPS_T) {
      float dx = p[i * 3 + 0] - cx;
      float dy = p[i * 3 + 1] - cy;
      float dz = p[i * 3 + 2] - cz;
      float d = dx * dx + dy * dy + dz * dz;
      float nd = fminf(db[i], d);
      db[i] = nd;
      if (nd > bv) { bv = nd; bi = i; }
    }
    sval[threadIdx.x] = bv; sidx[threadIdx.x] = bi;
    __syncthreads();
    for (int off = FPS_T / 2; off > 0; off >>= 1) {
      if (threadIdx.x < off) {
        if (sval[threadIdx.x + off] > sval[threadIdx.x]) {
          sval[threadIdx.x] = sval[threadIdx.x + off];
          sidx[threadIdx.x] = sidx[threadIdx.x + off];
        }
      }
      __syncthreads();
    }
    if (threadIdx.x == 0) sfar = sidx[0];
    __syncthreads();
  }
}

// ---------------------------------------------------------------------------
// Gather sampled centroids: new_xyz(B,P,3) = xyz[b, idx[b,p], :]
// ---------------------------------------------------------------------------
__global__ void gather_xyz_kernel(const float* __restrict__ xyz, const int* __restrict__ idx,
                                  float* __restrict__ out, int Np, int P, int total) {
  int t = blockIdx.x * blockDim.x + threadIdx.x;
  if (t >= total) return;
  int ch = t % 3;
  int bq = t / 3;
  int b  = bq / P;
  int gi = idx[bq];
  out[t] = xyz[((long)b * Np + gi) * 3 + ch];
}

// ---------------------------------------------------------------------------
// Ball query: thread per (b, query); in-order scan keeps indices sorted.
// ---------------------------------------------------------------------------
__global__ void ball_query_kernel(const float* __restrict__ xyz, const float* __restrict__ new_xyz,
                                  int* __restrict__ gidx, int Np, int np, int totq, float r2) {
  int t = blockIdx.x * blockDim.x + threadIdx.x;
  if (t >= totq) return;
  int b = t / np;
  const float* p = xyz + (long)b * Np * 3;
  float qx = new_xyz[t * 3 + 0], qy = new_xyz[t * 3 + 1], qz = new_xyz[t * 3 + 2];
  int* o = gidx + (long)t * NS;
  int cnt = 0, first = -1;
  for (int i = 0; i < Np && cnt < NS; ++i) {
    float dx = p[i * 3 + 0] - qx;
    float dy = p[i * 3 + 1] - qy;
    float dz = p[i * 3 + 2] - qz;
    float d = dx * dx + dy * dy + dz * dz;
    if (d <= r2) {
      if (first < 0) first = i;
      o[cnt++] = i;
    }
  }
  if (first < 0) first = 0;
  for (; cnt < NS; ++cnt) o[cnt] = first;
}

// ---------------------------------------------------------------------------
// Group + concat into padded f16: g(b,q,s, 0:3)=dxyz, (3:3+C)=feats, (3+C:Kp)=0
// ---------------------------------------------------------------------------
__global__ void group_concat_kernel(const float* __restrict__ xyz, const _Float16* __restrict__ feats,
                                    const float* __restrict__ new_xyz, const int* __restrict__ gidx,
                                    _Float16* __restrict__ g, int np, int Np, int C, int Kp,
                                    int total) {
  int t = blockIdx.x * blockDim.x + threadIdx.x;
  if (t >= total) return;
  int ch = t % Kp;
  int rest = t / Kp;
  int s  = rest % NS;
  int bq = rest / NS;
  int b  = bq / np;
  int gi = gidx[(long)bq * NS + s];
  float v;
  if (ch < 3)          v = xyz[((long)b * Np + gi) * 3 + ch] - new_xyz[(long)bq * 3 + ch];
  else if (ch < 3 + C) v = (float)feats[((long)b * Np + gi) * C + (ch - 3)];
  else                 v = 0.0f;
  g[t] = (_Float16)v;
}

// ---------------------------------------------------------------------------
// WMMA GEMM: C(M,N) f16 = relu(A(M,Kp) f16 @ Wt(N,Kp)^T f16 + bias f32).
// Kp % 32 == 0, N % 32 == 0, M % 128 == 0 -> branch-free, EXEC all-ones.
// One wave computes a 32x32 tile: 2 A-frags x 2 B-frags -> 4 WMMAs per
// K-chunk, 32 B of b128 loads per WMMA (register-level operand reuse).
// Fragment layouts follow CDNA5 ISA (16-bit A 16x32, f32 C 16x16).
// ---------------------------------------------------------------------------
__global__ void gemm_f16_wmma(const _Float16* __restrict__ A, const _Float16* __restrict__ Wt,
                              const float* __restrict__ bias, _Float16* __restrict__ C,
                              int M, int Kp, int N, int do_relu) {
  const int wave = threadIdx.x >> 5;
  const int lane = threadIdx.x & 31;
  const int m0 = blockIdx.x * 128 + wave * 32;
  const int n0 = blockIdx.y * 32;
  const int lr  = lane & 15;             // A-row / B-col within tile
  const int hi8 = (lane >> 4) * 8;       // half-wave K offset
  const _Float16* Ar0 = A  + (long)(m0 + lr) * Kp + hi8;
  const _Float16* Ar1 = A  + (long)(m0 + 16 + lr) * Kp + hi8;
  const _Float16* Bc0 = Wt + (long)(n0 + lr) * Kp + hi8;
  const _Float16* Bc1 = Wt + (long)(n0 + 16 + lr) * Kp + hi8;
  v8f acc00 = {}, acc01 = {}, acc10 = {}, acc11 = {};
  for (int kb = 0; kb < Kp; kb += 32) {
    V16 a0, a1, b0, b1;
    a0.h[0] = *(const h8*)(Ar0 + kb);
    a0.h[1] = *(const h8*)(Ar0 + kb + 16);
    a1.h[0] = *(const h8*)(Ar1 + kb);
    a1.h[1] = *(const h8*)(Ar1 + kb + 16);
    b0.h[0] = *(const h8*)(Bc0 + kb);
    b0.h[1] = *(const h8*)(Bc0 + kb + 16);
    b1.h[0] = *(const h8*)(Bc1 + kb);
    b1.h[1] = *(const h8*)(Bc1 + kb + 16);
    acc00 = __builtin_amdgcn_wmma_f32_16x16x32_f16(false, a0.v, false, b0.v, (short)0, acc00,
                                                   false, false);
    acc01 = __builtin_amdgcn_wmma_f32_16x16x32_f16(false, a0.v, false, b1.v, (short)0, acc01,
                                                   false, false);
    acc10 = __builtin_amdgcn_wmma_f32_16x16x32_f16(false, a1.v, false, b0.v, (short)0, acc10,
                                                   false, false);
    acc11 = __builtin_amdgcn_wmma_f32_16x16x32_f16(false, a1.v, false, b1.v, (short)0, acc11,
                                                   false, false);
  }
  const float bv0 = bias[n0 + lr];
  const float bv1 = bias[n0 + 16 + lr];
#pragma unroll
  for (int r = 0; r < 8; ++r) {
    int mA = m0 + r + (hi8 ? 8 : 0);
    int mB = mA + 16;
    float v00 = acc00[r] + bv0;
    float v01 = acc01[r] + bv1;
    float v10 = acc10[r] + bv0;
    float v11 = acc11[r] + bv1;
    if (do_relu) {
      v00 = fmaxf(v00, 0.0f); v01 = fmaxf(v01, 0.0f);
      v10 = fmaxf(v10, 0.0f); v11 = fmaxf(v11, 0.0f);
    }
    C[(long)mA * N + n0 + lr]      = (_Float16)v00;
    C[(long)mA * N + n0 + 16 + lr] = (_Float16)v01;
    C[(long)mB * N + n0 + lr]      = (_Float16)v10;
    C[(long)mB * N + n0 + 16 + lr] = (_Float16)v11;
  }
}

// ---------------------------------------------------------------------------
// Max-pool over nsample: out(B,P,C) = max_s g(B,P,S,C)   (f16 in/out)
// ---------------------------------------------------------------------------
__global__ void max_pool_kernel(const _Float16* __restrict__ g, _Float16* __restrict__ out,
                                int C, int total) {
  int t = blockIdx.x * blockDim.x + threadIdx.x;
  if (t >= total) return;
  int ch = t % C;
  long bq = t / C;
  const _Float16* base = g + ((long)bq * NS) * C + ch;
  float m = -1e30f;
  for (int s = 0; s < NS; ++s) m = fmaxf(m, (float)base[(long)s * C]);
  out[t] = (_Float16)m;
}

// ---------------------------------------------------------------------------
// 3-NN with normalized inverse-distance weights (brute force over n2)
// ---------------------------------------------------------------------------
__global__ void three_nn_kernel(const float* __restrict__ xyz1, const float* __restrict__ xyz2,
                                int* __restrict__ idx3, float* __restrict__ w3,
                                int n1, int n2, int total) {
  int t = blockIdx.x * blockDim.x + threadIdx.x;
  if (t >= total) return;
  int b = t / n1;
  float px = xyz1[(long)t * 3 + 0], py = xyz1[(long)t * 3 + 1], pz = xyz1[(long)t * 3 + 2];
  const float* q = xyz2 + (long)b * n2 * 3;
  float d0 = 1e30f, d1 = 1e30f, d2 = 1e30f;
  int i0 = 0, i1 = 0, i2 = 0;
  for (int j = 0; j < n2; ++j) {
    float dx = q[j * 3 + 0] - px;
    float dy = q[j * 3 + 1] - py;
    float dz = q[j * 3 + 2] - pz;
    float d = dx * dx + dy * dy + dz * dz;
    if (d < d0)      { d2 = d1; i2 = i1; d1 = d0; i1 = i0; d0 = d; i0 = j; }
    else if (d < d1) { d2 = d1; i2 = i1; d1 = d;  i1 = j; }
    else if (d < d2) { d2 = d;  i2 = j; }
  }
  float w0 = 1.0f / (d0 + 1e-8f), w1 = 1.0f / (d1 + 1e-8f), w2 = 1.0f / (d2 + 1e-8f);
  float s = w0 + w1 + w2;
  idx3[(long)t * 3 + 0] = i0; idx3[(long)t * 3 + 1] = i1; idx3[(long)t * 3 + 2] = i2;
  w3[(long)t * 3 + 0] = w0 / s; w3[(long)t * 3 + 1] = w1 / s; w3[(long)t * 3 + 2] = w2 / s;
}

// ---------------------------------------------------------------------------
// Interpolate + concat (f16): out(B,n1, C1+C2) = [p1 | sum_k w_k * p2[idx_k]]
// ---------------------------------------------------------------------------
__global__ void interp_concat_kernel(const _Float16* __restrict__ p1, const _Float16* __restrict__ p2,
                                     const int* __restrict__ idx3, const float* __restrict__ w3,
                                     _Float16* __restrict__ out, int n1, int n2,
                                     int C1, int C2, int total) {
  int t = blockIdx.x * blockDim.x + threadIdx.x;
  if (t >= total) return;
  int C = C1 + C2;
  int ch = t % C;
  int pt = t / C;
  int b  = pt / n1;
  float v;
  if (ch < C1) {
    v = (float)p1[(long)pt * C1 + ch];
  } else {
    int c = ch - C1;
    v = 0.0f;
    for (int k = 0; k < 3; ++k) {
      int j = idx3[(long)pt * 3 + k];
      v += w3[(long)pt * 3 + k] * (float)p2[((long)b * n2 + j) * C2 + c];
    }
  }
  out[t] = (_Float16)v;
}

// ---------------------------------------------------------------------------
// Final 128->13 conv + log_softmax (13 not a WMMA multiple -> scalar)
// ---------------------------------------------------------------------------
__global__ void conv_logsoftmax_kernel(const _Float16* __restrict__ f, const float* __restrict__ W,
                                       const float* __restrict__ bias, float* __restrict__ out,
                                       int total) {
  int t = blockIdx.x * blockDim.x + threadIdx.x;
  if (t >= total) return;
  const _Float16* x = f + (long)t * 128;
  float lg[CLS];
#pragma unroll
  for (int c = 0; c < CLS; ++c) lg[c] = bias[c];
  for (int k = 0; k < 128; ++k) {
    float xv = (float)x[k];
    const float* wr = W + k * CLS;
#pragma unroll
    for (int c = 0; c < CLS; ++c) lg[c] += xv * wr[c];
  }
  float mx = lg[0];
#pragma unroll
  for (int c = 1; c < CLS; ++c) mx = fmaxf(mx, lg[c]);
  float se = 0.0f;
#pragma unroll
  for (int c = 0; c < CLS; ++c) se += __expf(lg[c] - mx);
  float lse = mx + __logf(se);
#pragma unroll
  for (int c = 0; c < CLS; ++c) out[(long)t * CLS + c] = lg[c] - lse;
}

// ---------------------------------------------------------------------------
// Host orchestration
// ---------------------------------------------------------------------------
struct Layer { int widx, K, Kp, N; _Float16* Wt; };

extern "C" void kernel_launch(void* const* d_in, const int* in_sizes, int n_in,
                              void* d_out, int out_size, void* d_ws, size_t ws_size,
                              hipStream_t stream) {
  (void)in_sizes; (void)n_in; (void)out_size; (void)ws_size;

  const float* x = (const float*)d_in[0];
  // Weight index map (x, then sa1..sa4, fp4, fp3, fp2, fp1, conv; (W,b) pairs):
  // sa1:1..6  sa2:7..12  sa3:13..18  sa4:19..24  fp4:25..28 fp3:29..32
  // fp2:33..36 fp1:37..44 conv:45..46

  // ---- workspace bump allocator ----
  char* wsb = (char*)d_ws;
  size_t off = 0;
  auto allocf = [&](size_t n) { float* p = (float*)(wsb + off); off += ((n * 4 + 255) / 256) * 256; return p; };
  auto alloci = [&](size_t n) { int* p = (int*)(wsb + off); off += ((n * 4 + 255) / 256) * 256; return p; };
  auto alloch = [&](size_t n) { _Float16* p = (_Float16*)(wsb + off); off += ((n * 2 + 255) / 256) * 256; return p; };

  float*     c0 = allocf((size_t)Bc * Nc * 3);
  _Float16*  f0 = alloch((size_t)Bc * Nc * 6);
  float*     c1 = allocf((size_t)Bc * 1024 * 3);
  _Float16*  f1 = alloch((size_t)Bc * 1024 * 64);
  float*     c2 = allocf((size_t)Bc * 256 * 3);
  _Float16*  f2 = alloch((size_t)Bc * 256 * 128);
  float*     c3 = allocf((size_t)Bc * 64 * 3);
  _Float16*  f3 = alloch((size_t)Bc * 64 * 256);
  float*     c4 = allocf((size_t)Bc * 16 * 3);
  _Float16*  f4 = alloch((size_t)Bc * 16 * 512);
  _Float16* f3n = alloch((size_t)Bc * 64 * 256);
  _Float16* f2n = alloch((size_t)Bc * 256 * 256);
  _Float16* f1n = alloch((size_t)Bc * 1024 * 128);
  _Float16* f0n = alloch((size_t)Bc * Nc * 128);
  float* fpsdist = allocf((size_t)Bc * Nc);
  int*   sidx  = alloci((size_t)Bc * 1024);
  int*   gidx  = alloci((size_t)Bc * 1024 * NS);
  int*   nnidx = alloci((size_t)Bc * Nc * 3);
  float* nnw   = allocf((size_t)Bc * Nc * 3);
  const size_t ARENA = 8388608;   // halves; max(B*1024*32*64, B*N*128)
  _Float16* TA = alloch(ARENA);
  _Float16* TB = alloch(ARENA);
  _Float16* TC = alloch(ARENA);

  auto cdiv = [](long a, long b) { return (int)((a + b - 1) / b); };

  // ---- convert + transpose all MLP weights to f16 Wt(N,Kp) ----
  Layer L[20] = {
    {1, 9, 32, 32, nullptr},  {3, 32, 32, 32, nullptr},  {5, 32, 32, 64, nullptr},
    {7, 67, 96, 64, nullptr}, {9, 64, 64, 64, nullptr},  {11, 64, 64, 128, nullptr},
    {13, 131, 160, 128, nullptr}, {15, 128, 128, 128, nullptr}, {17, 128, 128, 256, nullptr},
    {19, 259, 288, 256, nullptr}, {21, 256, 256, 256, nullptr}, {23, 256, 256, 512, nullptr},
    {25, 768, 768, 256, nullptr}, {27, 256, 256, 256, nullptr},
    {29, 384, 384, 256, nullptr}, {31, 256, 256, 256, nullptr},
    {33, 320, 320, 256, nullptr}, {35, 256, 256, 128, nullptr},
    {37, 128, 128, 128, nullptr}, {39, 128, 128, 128, nullptr},
  };
  for (int i = 0; i < 20; ++i) {
    L[i].Wt = alloch((size_t)L[i].Kp * L[i].N);
    int tot = L[i].Kp * L[i].N;
    wconv_kernel<<<cdiv(tot, 256), 256, 0, stream>>>((const float*)d_in[L[i].widx], L[i].Wt,
                                                     L[i].K, L[i].Kp, L[i].N, tot);
  }
  // fp1 layers 3 and 4 (widx 41, 43) reuse the 128x128 shape; convert separately.
  _Float16* WtFp1c = alloch((size_t)128 * 128);
  _Float16* WtFp1d = alloch((size_t)128 * 128);
  {
    int tot = 128 * 128;
    wconv_kernel<<<cdiv(tot, 256), 256, 0, stream>>>((const float*)d_in[41], WtFp1c, 128, 128, 128, tot);
    wconv_kernel<<<cdiv(tot, 256), 256, 0, stream>>>((const float*)d_in[43], WtFp1d, 128, 128, 128, tot);
  }

  auto gemm = [&](const _Float16* A, const _Float16* Wt, int bidx, _Float16* Cb,
                  int M, int Kp, int Nd, bool relu) {
    const float* bp = (const float*)d_in[bidx];
    dim3 grid(M / 128, Nd / 32);
    gemm_f16_wmma<<<grid, dim3(128), 0, stream>>>(A, Wt, bp, Cb, M, Kp, Nd, relu ? 1 : 0);
  };

  // ---- split input ----
  {
    int tot = Bc * Nc * 9;
    split_kernel<<<cdiv(tot, 256), 256, 0, stream>>>(x, c0, f0, tot);
  }

  // ---- set abstraction stage ----
  auto sa_stage = [&](const float* xyz, int Np, const _Float16* feats, int Cf,
                      int P, float radius, int l0,
                      float* new_xyz, _Float16* new_feats) {
    fps_kernel<<<Bc, FPS_T, 0, stream>>>(xyz, sidx, fpsdist, Np, P);
    int tg = Bc * P * 3;
    gather_xyz_kernel<<<cdiv(tg, 256), 256, 0, stream>>>(xyz, sidx, new_xyz, Np, P, tg);
    int tq = Bc * P;
    ball_query_kernel<<<cdiv(tq, 256), 256, 0, stream>>>(xyz, new_xyz, gidx, Np, P, tq,
                                                         radius * radius);
    int Kp0 = L[l0].Kp;
    long tot = (long)Bc * P * NS * Kp0;
    group_concat_kernel<<<cdiv(tot, 256), 256, 0, stream>>>(xyz, feats, new_xyz, gidx, TA,
                                                            P, Np, Cf, Kp0, (int)tot);
    int R = Bc * P * NS;
    gemm(TA, L[l0].Wt,     L[l0].widx + 1,     TB, R, L[l0].Kp,     L[l0].N,     true);
    gemm(TB, L[l0 + 1].Wt, L[l0 + 1].widx + 1, TC, R, L[l0 + 1].Kp, L[l0 + 1].N, true);
    gemm(TC, L[l0 + 2].Wt, L[l0 + 2].widx + 1, TA, R, L[l0 + 2].Kp, L[l0 + 2].N, true);
    int h3 = L[l0 + 2].N;
    int tp = Bc * P * h3;
    max_pool_kernel<<<cdiv(tp, 256), 256, 0, stream>>>(TA, new_feats, h3, tp);
  };

  sa_stage(c0, Nc,   f0, 6,   1024, 0.1f, 0,  c1, f1);
  sa_stage(c1, 1024, f1, 64,  256,  0.2f, 3,  c2, f2);
  sa_stage(c2, 256,  f2, 128, 64,   0.4f, 6,  c3, f3);
  sa_stage(c3, 64,   f3, 256, 16,   0.8f, 9,  c4, f4);

  // ---- feature propagation stage ----
  auto fp_prep = [&](const float* xyz1, int n1, const float* xyz2, int n2,
                     const _Float16* p1, int C1, const _Float16* p2, int C2) {
    int t1 = Bc * n1;
    three_nn_kernel<<<cdiv(t1, 256), 256, 0, stream>>>(xyz1, xyz2, nnidx, nnw, n1, n2, t1);
    long tot = (long)t1 * (C1 + C2);
    interp_concat_kernel<<<cdiv(tot, 256), 256, 0, stream>>>(p1, p2, nnidx, nnw, TA,
                                                             n1, n2, C1, C2, (int)tot);
  };

  // fp4: 64 pts, [f3(256)|interp f4(512)] = 768 -> 256 -> 256
  fp_prep(c3, 64, c4, 16, f3, 256, f4, 512);
  gemm(TA, L[12].Wt, 26, TB, Bc * 64, 768, 256, true);
  gemm(TB, L[13].Wt, 28, f3n, Bc * 64, 256, 256, true);

  // fp3: 256 pts, [f2(128)|interp f3n(256)] = 384 -> 256 -> 256
  fp_prep(c2, 256, c3, 64, f2, 128, f3n, 256);
  gemm(TA, L[14].Wt, 30, TB, Bc * 256, 384, 256, true);
  gemm(TB, L[15].Wt, 32, f2n, Bc * 256, 256, 256, true);

  // fp2: 1024 pts, [f1(64)|interp f2n(256)] = 320 -> 256 -> 128
  fp_prep(c1, 1024, c2, 256, f1, 64, f2n, 256);
  gemm(TA, L[16].Wt, 34, TB, Bc * 1024, 320, 256, true);
  gemm(TB, L[17].Wt, 36, f1n, Bc * 1024, 256, 128, true);

  // fp1: 16384 pts, interp f1n(128) -> 128 x4
  fp_prep(c0, Nc, c1, 1024, nullptr, 0, f1n, 128);
  gemm(TA, L[18].Wt, 38, TB, Bc * Nc, 128, 128, true);
  gemm(TB, L[19].Wt, 40, TA, Bc * Nc, 128, 128, true);
  gemm(TA, WtFp1c,   42, TB, Bc * Nc, 128, 128, true);
  gemm(TB, WtFp1d,   44, f0n, Bc * Nc, 128, 128, true);

  // ---- final conv + log_softmax ----
  {
    const float* Wc = (const float*)d_in[45];
    const float* bcp = (const float*)d_in[46];
    int tot = Bc * Nc;
    conv_logsoftmax_kernel<<<cdiv(tot, 256), 256, 0, stream>>>(f0n, Wc, bcp, (float*)d_out, tot);
  }
}
